// DFTLayer_30837865185406
// MI455X (gfx1250) — matrix operand order repaired and checked by reading, not compiled, not run on hardware
//
#include <hip/hip_runtime.h>
#include <hip/hip_bf16.h>

// ---------------------------------------------------------------------------
// DFT-as-GEMM for MI455X (gfx1250, wave32, WMMA, async-to-LDS).
//   real[t,b,k] = sum_n frames[b,t,n] * wcos[k,n]
//  -imag[t,b,k] = sum_n frames[b,t,n] * (-wsin[k,n])
// Fused: C[m, j] = sum_n A[m,n] * WT[n,j],  m = b*2048+t,  j in [0,2048)
//   WT[n, j]      = bf16(wcos[j,n])            for j < 1024
//   WT[n, 1024+k] = bf16(-wsin[k,n])
// Output row index = t*16 + b (transpose folded into the store).
//
// Pipeline per K-step (BK=32):
//   store prefetched A regs -> LDS (f32->bf16)
//   s_wait_asynccnt 0            (B tile i landed in LDS via async DMA)
//   barrier
//   issue async B tile i+1 -> other LDS buffer; issue global A loads i+1
//   ds_load fragments, 8x v_wmma_f32_16x16x32_bf16   (hides the new loads)
//   barrier
// ---------------------------------------------------------------------------

#define N_FFT   1024
#define BATCH   16
#define TFRAMES 2048
#define FSTRIDE 1025
#define LTOT    (TFRAMES * FSTRIDE)        // 2099200 samples per batch row
#define MROWS   (BATCH * TFRAMES)          // 32768
#define NTOT    (2 * N_FFT)                // 2048 (cos | -sin)
#define HALF_OUT ((size_t)MROWS * N_FFT)   // offset of imag half in d_out

#define BM 128
#define BN 128
#define BK 32
#define KSTEPS (N_FFT / BK)                // 32
#define LDA_PITCH 40    // 32 + 8 pad (bf16 elems); row stride 80B (16B aligned)
#define LDB_PITCH 136   // 128 + 8 pad;            row stride 272B (16B aligned)

typedef __bf16 v16bf __attribute__((ext_vector_type(16)));
typedef float  v8f   __attribute__((ext_vector_type(8)));

union FragBF {
    uint4 u[2];     // 2 x 16B = 8 VGPRs
    v16bf v;        // 16 bf16 per lane
};

__device__ __forceinline__ unsigned short f2bf(float f) {
    unsigned u = __float_as_uint(f);
    unsigned r = (u + 0x7fffu + ((u >> 16) & 1u)) >> 16;   // RNE
    return (unsigned short)r;
}

// ---------------------------------------------------------------------------
// Prep: build WT (1024 x 2048 bf16) in workspace. One-time, 2M elements.
// ---------------------------------------------------------------------------
__global__ void dft_prep_weights(const float* __restrict__ wcos,
                                 const float* __restrict__ wsin,
                                 unsigned short* __restrict__ wt) {
    int idx = blockIdx.x * 256 + threadIdx.x;          // 0 .. 1024*2048-1
    if (idx >= N_FFT * NTOT) return;
    int n = idx >> 11;                                 // contraction index
    int j = idx & (NTOT - 1);                          // output column
    float v = (j < N_FFT) ? wcos[(size_t)j * N_FFT + n]
                          : -wsin[(size_t)(j - N_FFT) * N_FFT + n];
    wt[idx] = f2bf(v);                                 // idx == n*2048 + j
}

// ---------------------------------------------------------------------------
// Main GEMM: grid = (MROWS/BM = 256, NTOT/BN = 16), block = 256 (8 waves).
// Wave grid 4(M) x 2(N); each wave computes 32x64 via 2x4 wmma accumulators.
// ---------------------------------------------------------------------------
__global__ void __launch_bounds__(256)
dft_gemm_wmma(const float* __restrict__ x,
              const unsigned short* __restrict__ wt,
              float* __restrict__ out) {
    __shared__ __align__(16) unsigned short sA[BM * LDA_PITCH];        // 10240 B
    __shared__ __align__(16) unsigned short sB[2][BK * LDB_PITCH];     // 17408 B

    const int tid    = threadIdx.x;
    const int lane   = tid & 31;
    const int wave   = tid >> 5;
    const int wm     = wave & 3;    // 0..3  -> M sub-tile of 32 rows
    const int wn     = wave >> 2;   // 0..1  -> N sub-tile of 64 cols
    const int blockM = blockIdx.x * BM;
    const int jb     = blockIdx.y * BN;

    // Per-thread staging coordinates.
    const int ar = tid >> 3;              // A tile row   (0..31, +32 per pass)
    const int ac = (tid & 7) * 4;         // A tile col   (4 floats)
    const int br = tid >> 3;              // B tile row   (0..31)
    const int bc = (tid & 7) * 16;        // B tile col   (16 bf16)

    float areg[16];

    // Issue this thread's two async B128 copies: workspace -> LDS buffer `buf`.
    auto issueB = [&](int kk, int buf) {
        const unsigned short* g0 = wt + (size_t)(kk + br) * NTOT + jb + bc;
        unsigned l0 = (unsigned)(size_t)(const void*)&sB[buf][br * LDB_PITCH + bc];
        asm volatile("global_load_async_to_lds_b128 %0, %1, off"
                     :: "v"(l0), "v"(g0) : "memory");
        const unsigned short* g1 = g0 + 8;
        unsigned l1 = l0 + 16;
        asm volatile("global_load_async_to_lds_b128 %0, %1, off"
                     :: "v"(l1), "v"(g1) : "memory");
    };

    // Prefetch this thread's A-tile quad-rows for K-slice kk into registers.
    auto loadA = [&](int kk) {
        #pragma unroll
        for (int p = 0; p < 4; ++p) {
            int r = ar + p * 32;
            int m = blockM + r;
            int b = m >> 11;
            int t = m & (TFRAMES - 1);
            const float* src = x + (size_t)b * LTOT + (size_t)t * FSTRIDE + kk + ac;
            areg[4 * p + 0] = src[0];
            areg[4 * p + 1] = src[1];
            areg[4 * p + 2] = src[2];
            areg[4 * p + 3] = src[3];
        }
    };

    // Convert prefetched A regs to bf16 and publish to LDS.
    auto storeA = [&]() {
        #pragma unroll
        for (int p = 0; p < 4; ++p) {
            int r = ar + p * 32;
            __bf16* dst = (__bf16*)&sA[r * LDA_PITCH + ac];
            dst[0] = (__bf16)areg[4 * p + 0];
            dst[1] = (__bf16)areg[4 * p + 1];
            dst[2] = (__bf16)areg[4 * p + 2];
            dst[3] = (__bf16)areg[4 * p + 3];
        }
    };

    v8f acc[2][4] = {};

    // Prologue: tile 0 in flight.
    issueB(0, 0);
    loadA(0);

    for (int i = 0; i < KSTEPS; ++i) {
        const int buf = i & 1;

        storeA();                                          // A tile i -> LDS
        asm volatile("s_wait_asynccnt 0" ::: "memory");    // B tile i complete
        __syncthreads();

        if (i + 1 < KSTEPS) {                              // start tile i+1
            issueB((i + 1) * BK, buf ^ 1);
            loadA((i + 1) * BK);
        }

        // ---- fragments per documented CDNA5 VGPR layouts ----
        // A (16x32 bf16): lane -> M = lane%16; half = lane/16;
        //   elems 0..7 : K = 8*half + 0..7;  elems 8..15 : K = 8*half + 16..23
        FragBF afr[2];
        #pragma unroll
        for (int mi = 0; mi < 2; ++mi) {
            int row = wm * 32 + mi * 16 + (lane & 15);
            int kb  = (lane >> 4) * 8;
            const unsigned short* p = &sA[row * LDA_PITCH + kb];
            afr[mi].u[0] = *(const uint4*)(p);
            afr[mi].u[1] = *(const uint4*)(p + 16);
        }
        // B (32x16 bf16): lane -> N = lane%16; half = lane/16;
        //   elems 0..7 : K = 8*half + 0..7;  elems 8..15 : K = 8*half + 16..23
        FragBF bfr[4];
        #pragma unroll
        for (int ni = 0; ni < 4; ++ni) {
            const __bf16* p = (const __bf16*)&sB[buf][((lane >> 4) * 8) * LDB_PITCH + wn * 64 + ni * 16 + (lane & 15)];
            #pragma unroll
            for (int s = 0; s < 8; ++s) {
                bfr[ni].v[s]     = p[s * LDB_PITCH];
                bfr[ni].v[8 + s] = p[(16 + s) * LDB_PITCH];
            }
        }

        // ---- 8 WMMAs per wave per K-step ----
        #pragma unroll
        for (int mi = 0; mi < 2; ++mi)
            #pragma unroll
            for (int ni = 0; ni < 4; ++ni)
                acc[mi][ni] = __builtin_amdgcn_wmma_f32_16x16x32_bf16(
                    /*neg_a=*/false, afr[mi].v,
                    /*neg_b=*/false, bfr[ni].v,
                    /*c_mod=*/(short)0, acc[mi][ni],
                    /*reuse_a=*/false, /*reuse_b=*/false);

        __syncthreads();
    }

    // ---- epilogue: C/D layout lane -> N = lane%16, VGPR v -> M = v + 8*(lane/16)
    const bool  isImag = (jb >= N_FFT);
    float*      obase  = out + (isImag ? HALF_OUT : 0);
    const int   jcol0  = (isImag ? jb - N_FFT : jb) + wn * 64;

    #pragma unroll
    for (int mi = 0; mi < 2; ++mi) {
        int rowTile = blockM + wm * 32 + mi * 16 + ((lane >> 4) << 3);
        #pragma unroll
        for (int ni = 0; ni < 4; ++ni) {
            int j = jcol0 + ni * 16 + (lane & 15);
            #pragma unroll
            for (int v = 0; v < 8; ++v) {
                int m = rowTile + v;
                int b = m >> 11;
                int t = m & (TFRAMES - 1);
                obase[(((size_t)(t << 4) + b) << 10) + j] = acc[mi][ni][v];
            }
        }
    }
}

// ---------------------------------------------------------------------------
extern "C" void kernel_launch(void* const* d_in, const int* in_sizes, int n_in,
                              void* d_out, int out_size, void* d_ws, size_t ws_size,
                              hipStream_t stream) {
    (void)in_sizes; (void)n_in; (void)out_size; (void)ws_size;
    const float* x    = (const float*)d_in[0];
    const float* wsin = (const float*)d_in[1];
    const float* wcos = (const float*)d_in[2];
    unsigned short* wt = (unsigned short*)d_ws;   // 1024*2048 bf16 = 4 MB
    float* out = (float*)d_out;

    // Build fused bf16 weight matrix (deterministic, rebuilt every call).
    dft_prep_weights<<<(N_FFT * NTOT + 255) / 256, 256, 0, stream>>>(wcos, wsin, wt);

    // Fused real+imag GEMM.
    dim3 grid(MROWS / BM, NTOT / BN);   // 256 x 16
    dft_gemm_wmma<<<grid, 256, 0, stream>>>(x, wt, out);
}
